// CrossAttentionLayer_40793599378095
// MI455X (gfx1250) — matrix operand order repaired and checked by reading, not compiled, not run on hardware
//
#include <hip/hip_runtime.h>
#include <hip/hip_bf16.h>

// ---------------------------------------------------------------------------
// CDNA5 (gfx1250) sigmoid cross-attention, FP16 WMMA (f32 accumulate).
// Round 3: GEMM density bump — each wave owns 2 A-tiles (32 M rows) so every
// B-fragment feeds 2 WMMAs (12 loads -> 8 WMMAs per k-step), giving the
// scheduler enough independent WMMAs to keep the double buffer alive instead
// of collapsing to s_wait_loadcnt 0 before every matrix op.
// ---------------------------------------------------------------------------

typedef __attribute__((ext_vector_type(16))) _Float16 v16h;
typedef __attribute__((ext_vector_type(8)))  _Float16 v8h;
typedef __attribute__((ext_vector_type(4)))  _Float16 v4h;
typedef __attribute__((ext_vector_type(8)))  float    v8f;

#define D_MODEL   1024
#define NUM_HEAD  16
#define D_K       64
#define NQ        512
#define NK        1024
#define BS        8

// ---- WMMA wrapper: D = A(16x32 f16) * B(32x16 f16) + C(16x16 f32) ----------
__device__ __forceinline__ v8f wmma16(v16h a, v16h b, v8f c) {
    return __builtin_amdgcn_wmma_f32_16x16x32_f16(
        /*neg_a=*/false, a, /*neg_b=*/false, b,
        /*c_mod=*/(short)0, c, /*reuse_a=*/false, /*reuse_b=*/false);
}

// ---- A-fragment (16x32, M in lanes): lane<16 -> K {0..7,16..23}, lane>=16 ->
// K {8..15,24..31}.  Two 16-byte loads per lane.
__device__ __forceinline__ v16h ld_fragA(const _Float16* p, int stride) {
    const int lane = threadIdx.x & 31;
    const int hf   = lane >> 4;
    const _Float16* r = p + (lane & 15) * stride;
    const v8h lo = *(const v8h*)(r + hf * 8);
    const v8h hi = *(const v8h*)(r + 16 + hf * 8);
    v16h o;
#pragma unroll
    for (int i = 0; i < 8; ++i) { o[i] = lo[i]; o[i + 8] = hi[i]; }
    return o;
}

// ---- B-fragment (32x16, N in lanes): lane<16 -> K {0..7,16..23}, lane>=16 -> K {8..15,24..31}
// (same as A).  Column n of B must be contiguous in memory at p + n*stride.
__device__ __forceinline__ v16h ld_fragB(const _Float16* p, int stride) {
    const int lane = threadIdx.x & 31;
    const int hf   = lane >> 4;
    const _Float16* r = p + (lane & 15) * stride;
    const v8h lo = *(const v8h*)(r + hf * 8);
    const v8h hi = *(const v8h*)(r + 16 + hf * 8);
    v16h o;
#pragma unroll
    for (int i = 0; i < 8; ++i) { o[i] = lo[i]; o[i + 8] = hi[i]; }
    return o;
}

// ---------------------------------------------------------------------------
// LayerNorm over last dim (1024) -> f16.  One block per row, 4 elems/thread.
// ---------------------------------------------------------------------------
__global__ __launch_bounds__(256)
void ln_f16_kernel(const float* __restrict__ X, const float* __restrict__ w,
                   const float* __restrict__ beta, _Float16* __restrict__ Y) {
    __shared__ float rs[256];
    __shared__ float rq[256];
    const int tid = threadIdx.x;
    const size_t row = blockIdx.x;
    const float4 x = ((const float4*)(X + row * D_MODEL))[tid];
    rs[tid] = x.x + x.y + x.z + x.w;
    rq[tid] = x.x * x.x + x.y * x.y + x.z * x.z + x.w * x.w;
    __syncthreads();
#pragma unroll
    for (int off = 128; off > 0; off >>= 1) {
        if (tid < off) { rs[tid] += rs[tid + off]; rq[tid] += rq[tid + off]; }
        __syncthreads();
    }
    const float mu   = rs[0] * (1.0f / D_MODEL);
    const float var  = rq[0] * (1.0f / D_MODEL) - mu * mu;
    const float rstd = rsqrtf(var + 1e-5f);
    const float4 wv = ((const float4*)w)[tid];
    const float4 bv = ((const float4*)beta)[tid];
    v4h y;
    y[0] = (_Float16)((x.x - mu) * rstd * wv.x + bv.x);
    y[1] = (_Float16)((x.y - mu) * rstd * wv.y + bv.y);
    y[2] = (_Float16)((x.z - mu) * rstd * wv.z + bv.z);
    y[3] = (_Float16)((x.w - mu) * rstd * wv.w + bv.w);
    ((v4h*)(Y + row * D_MODEL))[tid] = y;
}

// ---------------------------------------------------------------------------
// f32 -> f16 weight convert (1M elements, 4 per thread)
// ---------------------------------------------------------------------------
__global__ __launch_bounds__(256)
void cvt_f16_kernel(const float* __restrict__ in, _Float16* __restrict__ out) {
    const int i = (blockIdx.x * 256 + threadIdx.x);
    const float4 x = ((const float4*)in)[i];
    v4h y;
    y[0] = (_Float16)x.x; y[1] = (_Float16)x.y;
    y[2] = (_Float16)x.z; y[3] = (_Float16)x.w;
    ((v4h*)out)[i] = y;
}

// ---------------------------------------------------------------------------
// Y[M x 1024] = X[M x 1024] @ W^T + bias, f16 in / f16 out, f32 accum.
// grid = (M/256, 1024/64), 256 threads (8 waves).  Wave w owns 32 M-rows
// (two 16-row A-tiles sharing the same B-fragments).  Software-pipelined:
// k-step i+1 fragments are issued before the 8 WMMAs consuming step i.
// ---------------------------------------------------------------------------
__global__ __launch_bounds__(256)
void gemm_xwT_kernel(const _Float16* __restrict__ X, const _Float16* __restrict__ Wh,
                     const float* __restrict__ bias, _Float16* __restrict__ Y) {
    const int wave = threadIdx.x >> 5;
    const int lane = threadIdx.x & 31;
    const int hf   = lane >> 4;
    const int ln   = lane & 15;
    const int mbase = blockIdx.x * 256 + wave * 32;
    const int nbase = blockIdx.y * 64;

    v8f acc[2][4];
#pragma unroll
    for (int mt = 0; mt < 2; ++mt)
#pragma unroll
        for (int nt = 0; nt < 4; ++nt) acc[mt][nt] = (v8f)(0.0f);

    const _Float16* xp0 = X + (size_t)mbase * D_MODEL;
    const _Float16* xp1 = xp0 + (size_t)16 * D_MODEL;
    const _Float16* wp  = Wh + (size_t)nbase * D_MODEL;

    // prologue: k-step 0 fragments
    v16h a_c[2];
    a_c[0] = ld_fragA(xp0, D_MODEL);
    a_c[1] = ld_fragA(xp1, D_MODEL);
    v16h b_c[4];
#pragma unroll
    for (int nt = 0; nt < 4; ++nt)
        b_c[nt] = ld_fragB(wp + (size_t)(nt * 16) * D_MODEL, D_MODEL);

#pragma unroll 2
    for (int kk = 32; kk < D_MODEL; kk += 32) {
        // issue next-step loads first
        v16h a_n[2];
        a_n[0] = ld_fragA(xp0 + kk, D_MODEL);
        a_n[1] = ld_fragA(xp1 + kk, D_MODEL);
        v16h b_n[4];
#pragma unroll
        for (int nt = 0; nt < 4; ++nt)
            b_n[nt] = ld_fragB(wp + (size_t)(nt * 16) * D_MODEL + kk, D_MODEL);
        // consume current-step fragments: 8 independent WMMAs
#pragma unroll
        for (int nt = 0; nt < 4; ++nt) {
            acc[0][nt] = wmma16(a_c[0], b_c[nt], acc[0][nt]);
            acc[1][nt] = wmma16(a_c[1], b_c[nt], acc[1][nt]);
        }
        a_c[0] = a_n[0];
        a_c[1] = a_n[1];
#pragma unroll
        for (int nt = 0; nt < 4; ++nt) b_c[nt] = b_n[nt];
    }
    // epilogue k-step
#pragma unroll
    for (int nt = 0; nt < 4; ++nt) {
        acc[0][nt] = wmma16(a_c[0], b_c[nt], acc[0][nt]);
        acc[1][nt] = wmma16(a_c[1], b_c[nt], acc[1][nt]);
    }

#pragma unroll
    for (int mt = 0; mt < 2; ++mt) {
#pragma unroll
        for (int nt = 0; nt < 4; ++nt) {
            const float bn = bias[nbase + nt * 16 + ln];
#pragma unroll
            for (int r = 0; r < 8; ++r) {
                const int m = mt * 16 + r + hf * 8;
                Y[(size_t)(mbase + m) * D_MODEL + nbase + nt * 16 + ln] =
                    (_Float16)(acc[mt][nt][r] + bn);
            }
        }
    }
}

// ---------------------------------------------------------------------------
// Sigmoid cross-attention for one (b, h, 128-q tile).
// 256 threads = 8 waves; wave w owns q rows [qt*128 + w*16, +16).
// ---------------------------------------------------------------------------
#define SSTR 136          // padded LDS k-stride (keeps 16B alignment, odd*8)

__global__ __launch_bounds__(256)
void attn_kernel(const _Float16* __restrict__ Q, const _Float16* __restrict__ K,
                 const _Float16* __restrict__ V, float* __restrict__ out,
                 float* __restrict__ attn_mean) {
    __shared__ _Float16 sVt[D_K * SSTR];       // V chunk, transposed [d][k]
    __shared__ _Float16 sS[8 * 16 * SSTR];     // per-wave sigmoid(S) [m][k]

    const int qt   = blockIdx.x;               // 0..3
    const int b    = blockIdx.y >> 4;
    const int h    = blockIdx.y & 15;
    const int wave = threadIdx.x >> 5;
    const int lane = threadIdx.x & 31;
    const int hf   = lane >> 4;
    const int ln   = lane & 15;
    const int qbase = qt * 128 + wave * 16;

    // Q A-fragments for this wave's 16 rows (d 0..31 and 32..63), kept live.
    const _Float16* qp = Q + (size_t)(b * NQ + qbase) * D_MODEL + h * D_K;
    const v16h a0 = ld_fragA(qp, D_MODEL);
    const v16h a1 = ld_fragA(qp + 32, D_MODEL);

    v8f o[4];
#pragma unroll
    for (int t = 0; t < 4; ++t) o[t] = (v8f)(0.0f);

    _Float16* mySS = sS + wave * 16 * SSTR;

#pragma unroll 1
    for (int kc = 0; kc < NK / 128; ++kc) {
        // ---- prefetch next chunk's K and V rows (global_prefetch_b8) ------
        if (kc + 1 < NK / 128) {
            const int prow = threadIdx.x & 127;
            const _Float16* np = (threadIdx.x < 128)
                ? K + (size_t)(b * NK + (kc + 1) * 128 + prow) * D_MODEL + h * D_K
                : V + (size_t)(b * NK + (kc + 1) * 128 + prow) * D_MODEL + h * D_K;
            __builtin_prefetch(np, 0, 0);
        }

        // ---- stage V chunk transposed: sVt[d][k], k local in [0,128) ------
        {
            const int krow = threadIdx.x >> 1;          // 0..127
            const int cb   = (threadIdx.x & 1) * 32;    // d 0..31 / 32..63
            const _Float16* vp =
                V + (size_t)(b * NK + kc * 128 + krow) * D_MODEL + h * D_K + cb;
            const v8h v0 = *(const v8h*)(vp);
            const v8h v1 = *(const v8h*)(vp + 8);
            const v8h v2 = *(const v8h*)(vp + 16);
            const v8h v3 = *(const v8h*)(vp + 24);
#pragma unroll
            for (int c = 0; c < 8; ++c) {
                sVt[(cb + c) * SSTR + krow]      = v0[c];
                sVt[(cb + 8 + c) * SSTR + krow]  = v1[c];
                sVt[(cb + 16 + c) * SSTR + krow] = v2[c];
                sVt[(cb + 24 + c) * SSTR + krow] = v3[c];
            }
        }
        __syncthreads();

        // ---- phase 1: S tiles, sigmoid, head-mean atomics, stash in LDS ---
        const _Float16* kp = K + (size_t)(b * NK + kc * 128) * D_MODEL + h * D_K;
        v16h b0c = ld_fragB(kp, D_MODEL);
        v16h b1c = ld_fragB(kp + 32, D_MODEL);
#pragma unroll
        for (int t = 0; t < 8; ++t) {
            v16h b0n = b0c, b1n = b1c;
            if (t < 7) {
                const _Float16* kt = kp + (size_t)((t + 1) * 16) * D_MODEL;
                b0n = ld_fragB(kt, D_MODEL);
                b1n = ld_fragB(kt + 32, D_MODEL);
            }
            v8f s = (v8f)(0.0f);
            s = wmma16(a0, b0c, s);
            s = wmma16(a1, b1c, s);
#pragma unroll
            for (int r = 0; r < 8; ++r) {
                const float x  = s[r] * 0.125f;               // 1/sqrt(64)
                const float sg = __builtin_amdgcn_rcpf(
                    1.0f + __builtin_amdgcn_exp2f(-1.4426950408889634f * x));
                const int m  = r + hf * 8;
                const int kg = kc * 128 + t * 16 + ln;
                atomicAdd(&attn_mean[(size_t)(b * NQ + qbase + m) * NK + kg],
                          sg * (1.0f / NUM_HEAD));
                mySS[m * SSTR + t * 16 + ln] = (_Float16)sg;
            }
            b0c = b0n;
            b1c = b1n;
        }
        // wave-private LDS store->load turnaround
        asm volatile("s_wait_dscnt 0" ::: "memory");

        // ---- phase 2: O += sigmoid(S) @ V ---------------------------------
        v16h sa_c = ld_fragA(mySS, SSTR);
#pragma unroll
        for (int ks = 0; ks < 4; ++ks) {
            v16h vb[4];
#pragma unroll
            for (int nt = 0; nt < 4; ++nt)
                vb[nt] = ld_fragB(sVt + (nt * 16) * SSTR + ks * 32, SSTR);
            v16h sa_n = sa_c;
            if (ks < 3) sa_n = ld_fragA(mySS + (ks + 1) * 32, SSTR);
#pragma unroll
            for (int nt = 0; nt < 4; ++nt)
                o[nt] = wmma16(sa_c, vb[nt], o[nt]);
            sa_c = sa_n;
        }
        __syncthreads();   // protect sVt before next chunk restages
    }

    // ---- epilogue: f32 out ------------------------------------------------
    float* op = out + (size_t)(b * NQ + qbase) * D_MODEL + h * D_K;
#pragma unroll
    for (int nt = 0; nt < 4; ++nt) {
#pragma unroll
        for (int r = 0; r < 8; ++r) {
            const int m = r + hf * 8;
            op[(size_t)m * D_MODEL + nt * 16 + ln] = o[nt][r];
        }
    }
}

// ---------------------------------------------------------------------------
extern "C" void kernel_launch(void* const* d_in, const int* in_sizes, int n_in,
                              void* d_out, int out_size, void* d_ws, size_t ws_size,
                              hipStream_t stream) {
    const float* text = (const float*)d_in[0];
    const float* av   = (const float*)d_in[1];
    const float* tn_w = (const float*)d_in[2];
    const float* tn_b = (const float*)d_in[3];
    const float* an_w = (const float*)d_in[4];
    const float* an_b = (const float*)d_in[5];
    const float* Wq   = (const float*)d_in[6];
    const float* bq   = (const float*)d_in[7];
    const float* Wk   = (const float*)d_in[8];
    const float* bk   = (const float*)d_in[9];
    const float* Wv   = (const float*)d_in[10];
    const float* bv   = (const float*)d_in[11];

    float* out = (float*)d_out;                       // (8,512,1024)
    float* am  = out + (size_t)BS * NQ * D_MODEL;     // attn.mean(-1): (8,512,1024)

    // workspace carve-up (all f16): 70 MB total
    _Float16* t_h  = (_Float16*)d_ws;                 // 4096*1024
    _Float16* a_h  = t_h  + (size_t)BS * NQ * D_MODEL;  // 8192*1024
    _Float16* wq_h = a_h  + (size_t)BS * NK * D_MODEL;
    _Float16* wk_h = wq_h + (size_t)D_MODEL * D_MODEL;
    _Float16* wv_h = wk_h + (size_t)D_MODEL * D_MODEL;
    _Float16* q_h  = wv_h + (size_t)D_MODEL * D_MODEL;  // 4096*1024
    _Float16* k_h  = q_h  + (size_t)BS * NQ * D_MODEL;  // 8192*1024
    _Float16* v_h  = k_h  + (size_t)BS * NK * D_MODEL;  // 8192*1024

    // attn_mean accumulated with atomics -> zero it each call
    hipMemsetAsync(am, 0, (size_t)BS * NQ * NK * sizeof(float), stream);

    // LayerNorms -> f16
    ln_f16_kernel<<<BS * NQ, 256, 0, stream>>>(text, tn_w, tn_b, t_h);
    ln_f16_kernel<<<BS * NK, 256, 0, stream>>>(av, an_w, an_b, a_h);

    // weights -> f16 (1M elems each, 4/thread)
    cvt_f16_kernel<<<D_MODEL * D_MODEL / (256 * 4), 256, 0, stream>>>(Wq, wq_h);
    cvt_f16_kernel<<<D_MODEL * D_MODEL / (256 * 4), 256, 0, stream>>>(Wk, wk_h);
    cvt_f16_kernel<<<D_MODEL * D_MODEL / (256 * 4), 256, 0, stream>>>(Wv, wv_h);

    // projections (WMMA): wave tile 32Mx64N, block tile 256Mx64N
    gemm_xwT_kernel<<<dim3(BS * NQ / 256, D_MODEL / 64), 256, 0, stream>>>(t_h, wq_h, bq, q_h);
    gemm_xwT_kernel<<<dim3(BS * NK / 256, D_MODEL / 64), 256, 0, stream>>>(a_h, wk_h, bk, k_h);
    gemm_xwT_kernel<<<dim3(BS * NK / 256, D_MODEL / 64), 256, 0, stream>>>(a_h, wv_h, bv, v_h);

    // sigmoid cross-attention (WMMA)
    attn_kernel<<<dim3(NQ / 128, BS * NUM_HEAD), 256, 0, stream>>>(q_h, k_h, v_h, out, am);
}
